// GAT_45449343926515
// MI455X (gfx1250) — matrix operand order repaired and checked by reading, not compiled, not run on hardware
//
#include <hip/hip_runtime.h>
#include <hip/hip_bf16.h>
#include <math.h>

typedef __attribute__((ext_vector_type(16))) _Float16 v16h;
typedef __attribute__((ext_vector_type(8)))  float    v8f;

#define NEG_SLOPE 0.2f
#define NEG_INF_F (-3.402823466e38f)

__device__ __forceinline__ float lrelu(float v) { return v > 0.0f ? v : NEG_SLOPE * v; }

// float atomic max via signed-max (positive) / unsigned-min (negative) bit trick.
// Correct for mixed-sign concurrent updates with -inf init.
__device__ __forceinline__ void atomicMaxFloat(float* addr, float val) {
  if (val >= 0.0f) atomicMax((int*)addr, __float_as_int(val));
  else             atomicMin((unsigned int*)addr, __float_as_uint(val));
}

// ---------------------------------------------------------------- init
__global__ void k_init(float* o1, float* mx1, float* dn1,
                       float* mx2, float* dn2, float* as2, float* ad2,
                       float* out, int N) {
  int i = blockIdx.x * blockDim.x + threadIdx.x;
  if (i >= N * 64) return;
  o1[i] = 0.0f;
  if (i < N * 40) out[i] = 0.0f;
  if (i < N * 8) { mx1[i] = NEG_INF_F; dn1[i] = 0.0f; }
  if (i < N)     { mx2[i] = NEG_INF_F; dn2[i] = 0.0f; as2[i] = 0.0f; ad2[i] = 0.0f; }
}

// ---------------------------------------------------------------- weight prep:
// W1 [512,64] f32 row-major -> W1t f16 column-major (col*512 + k)
// W2 [64,40]  f32 row-major -> W2t f16 column-major padded to 48 cols (col*64 + k)
// Each B fragment then becomes one 32-byte contiguous v16h load per lane.
__global__ void k_prep_w(const float* __restrict__ W1, const float* __restrict__ W2,
                         _Float16* __restrict__ W1t, _Float16* __restrict__ W2t) {
  int i = blockIdx.x * blockDim.x + threadIdx.x;
  if (i < 64 * 512) {
    int col = i >> 9, k = i & 511;
    W1t[i] = (_Float16)W1[k * 64 + col];
  }
  if (i < 48 * 64) {
    int col = i >> 6, k = i & 63;
    W2t[i] = (_Float16)(col < 40 ? W2[k * 40 + col] : 0.0f);
  }
}

// ---------------------------------------------------------------- GEMM1: h1 = x @ W1  [N,512]x[512,64]
// One wave per 16-row M tile; all 4 N-tiles accumulated in-wave so each x slab is
// streamed from HBM exactly once (x = 204.8MB > L2, so redundancy would be real traffic).
// Epilogue: per-(row,head) attention logits via 8-lane shfl reductions.
__global__ __launch_bounds__(256) void k_gemm1(
    const float* __restrict__ x, const _Float16* __restrict__ W1t,
    const float* __restrict__ aw_s, const float* __restrict__ aw_d,
    float* __restrict__ h1, float* __restrict__ as1, float* __restrict__ ad1, int N)
{
  int wave = (int)(((unsigned)blockIdx.x * blockDim.x + threadIdx.x) >> 5);
  int lane = threadIdx.x & 31;
  int mtiles = (N + 15) >> 4;
  if (wave >= mtiles) return;
  int half = lane >> 4, l15 = lane & 15;
  int mbase = wave << 4;
  int arow = mbase + l15; if (arow > N - 1) arow = N - 1;   // clamp: no in-loop branch
  const float* ap = x + (size_t)arow * 512 + half * 16;

  v8f acc0 = {}, acc1 = {}, acc2 = {}, acc3 = {};
  for (int k = 0; k < 512; k += 32) {
    // A fragment: lane holds row (lane&15), K = {half*16..+15}
    const float* pa = ap + k;
    float4 f0 = ((const float4*)pa)[0];
    float4 f1 = ((const float4*)pa)[1];
    float4 f2 = ((const float4*)(pa + 8))[0];
    float4 f3 = ((const float4*)(pa + 8))[1];
    v16h a;
    a[0]=(_Float16)f0.x;  a[1]=(_Float16)f0.y;  a[2]=(_Float16)f0.z;  a[3]=(_Float16)f0.w;
    a[4]=(_Float16)f1.x;  a[5]=(_Float16)f1.y;  a[6]=(_Float16)f1.z;  a[7]=(_Float16)f1.w;
    a[8]=(_Float16)f2.x;  a[9]=(_Float16)f2.y;  a[10]=(_Float16)f2.z; a[11]=(_Float16)f2.w;
    a[12]=(_Float16)f3.x; a[13]=(_Float16)f3.y; a[14]=(_Float16)f3.z; a[15]=(_Float16)f3.w;
    // B fragments: lane holds col (lane&15), 16 consecutive K starting at k + half*16
    const _Float16* pb = W1t + (size_t)l15 * 512 + k + half * 16;
    v16h b0 = *(const v16h*)(pb);
    v16h b1 = *(const v16h*)(pb + 16 * 512);
    v16h b2 = *(const v16h*)(pb + 32 * 512);
    v16h b3 = *(const v16h*)(pb + 48 * 512);
    acc0 = __builtin_amdgcn_wmma_f32_16x16x32_f16(false, a, false, b0, (short)0, acc0, false, false);
    acc1 = __builtin_amdgcn_wmma_f32_16x16x32_f16(false, a, false, b1, (short)0, acc1, false, false);
    acc2 = __builtin_amdgcn_wmma_f32_16x16x32_f16(false, a, false, b2, (short)0, acc2, false, false);
    acc3 = __builtin_amdgcn_wmma_f32_16x16x32_f16(false, a, false, b3, (short)0, acc3, false, false);
  }

  auto epi = [&](const v8f& A, int nb) {
    #pragma unroll
    for (int r = 0; r < 8; ++r) {
      int row = mbase + r + half * 8;
      int col = nb + l15;
      float v = A[r];
      bool rv = row < N;
      if (rv) h1[(size_t)row * 64 + col] = v;
      float s = v * aw_s[col];
      float d = v * aw_d[col];
      s += __shfl_xor(s, 1); s += __shfl_xor(s, 2); s += __shfl_xor(s, 4);
      d += __shfl_xor(d, 1); d += __shfl_xor(d, 2); d += __shfl_xor(d, 4);
      if (rv && (l15 & 7) == 0) {   // 8 cols == one full head -> complete dot product
        int head = col >> 3;
        as1[(size_t)row * 8 + head] = s;
        ad1[(size_t)row * 8 + head] = d;
      }
    }
  };
  epi(acc0, 0); epi(acc1, 16); epi(acc2, 32); epi(acc3, 48);
}

// ---------------------------------------------------------------- layer-1 edge passes (H=8)
__global__ void k_edge_max1(const int* __restrict__ src, const int* __restrict__ dst,
                            const float* __restrict__ as1, const float* __restrict__ ad1,
                            float* __restrict__ mx1, int E, int Ne) {
  int e = blockIdx.x * blockDim.x + threadIdx.x;
  if (e >= Ne) return;
  int s, d; if (e < E) { s = src[e]; d = dst[e]; } else { s = e - E; d = s; }
  const float4* ps = (const float4*)(as1 + (size_t)s * 8);
  const float4* pd = (const float4*)(ad1 + (size_t)d * 8);
  float4 s0 = ps[0], s1 = ps[1], d0 = pd[0], d1 = pd[1];
  float v[8] = { s0.x+d0.x, s0.y+d0.y, s0.z+d0.z, s0.w+d0.w,
                 s1.x+d1.x, s1.y+d1.y, s1.z+d1.z, s1.w+d1.w };
  float* m = mx1 + (size_t)d * 8;
  #pragma unroll
  for (int h = 0; h < 8; ++h) atomicMaxFloat(m + h, lrelu(v[h]));
}

__global__ void k_edge_den1(const int* __restrict__ src, const int* __restrict__ dst,
                            const float* __restrict__ as1, const float* __restrict__ ad1,
                            const float* __restrict__ mx1, float* __restrict__ dn1,
                            int E, int Ne) {
  int e = blockIdx.x * blockDim.x + threadIdx.x;
  if (e >= Ne) return;
  int s, d; if (e < E) { s = src[e]; d = dst[e]; } else { s = e - E; d = s; }
  const float4* ps = (const float4*)(as1 + (size_t)s * 8);
  const float4* pd = (const float4*)(ad1 + (size_t)d * 8);
  float4 s0 = ps[0], s1 = ps[1], d0 = pd[0], d1 = pd[1];
  float v[8] = { s0.x+d0.x, s0.y+d0.y, s0.z+d0.z, s0.w+d0.w,
                 s1.x+d1.x, s1.y+d1.y, s1.z+d1.z, s1.w+d1.w };
  const float* m = mx1 + (size_t)d * 8;
  float* dn = dn1 + (size_t)d * 8;
  #pragma unroll
  for (int h = 0; h < 8; ++h) atomicAdd(dn + h, __expf(lrelu(v[h]) - m[h]));
}

// one thread per (edge, head): gathers 8-ch message, scatters via L2-resident f32 atomics
__global__ void k_edge_msg1(const int* __restrict__ src, const int* __restrict__ dst,
                            const float* __restrict__ as1, const float* __restrict__ ad1,
                            const float* __restrict__ mx1, const float* __restrict__ dn1,
                            const float* __restrict__ h1, float* __restrict__ o1,
                            int E, int Ne) {
  int tid = blockIdx.x * blockDim.x + threadIdx.x;
  int e = tid >> 3, h = tid & 7;
  if (e >= Ne) return;
  int s, d; if (e < E) { s = src[e]; d = dst[e]; } else { s = e - E; d = s; }
  float v = lrelu(as1[(size_t)s * 8 + h] + ad1[(size_t)d * 8 + h]);
  float alpha = __expf(v - mx1[(size_t)d * 8 + h]) / (dn1[(size_t)d * 8 + h] + 1e-16f);
  const float4* hp = (const float4*)(h1 + (size_t)s * 64 + h * 8);
  float4 p0 = hp[0], p1 = hp[1];
  float* op = o1 + (size_t)d * 64 + h * 8;
  atomicAdd(op + 0, alpha * p0.x); atomicAdd(op + 1, alpha * p0.y);
  atomicAdd(op + 2, alpha * p0.z); atomicAdd(op + 3, alpha * p0.w);
  atomicAdd(op + 4, alpha * p1.x); atomicAdd(op + 5, alpha * p1.y);
  atomicAdd(op + 6, alpha * p1.z); atomicAdd(op + 7, alpha * p1.w);
}

__global__ void k_bias_elu(float* __restrict__ o1, const float* __restrict__ b1, int N) {
  int i = blockIdx.x * blockDim.x + threadIdx.x;
  if (i >= N * 64) return;
  float v = o1[i] + b1[i & 63];
  o1[i] = v > 0.0f ? v : (__expf(v) - 1.0f);
}

// ---------------------------------------------------------------- GEMM2: h2 = elu_out @ W2 [N,64]x[64,40->48]
// One wave per M tile, 3 N-tiles in-wave. W2t cols 40..47 are zero-padded: no load guard needed.
__global__ __launch_bounds__(256) void k_gemm2(
    const float* __restrict__ hin, const _Float16* __restrict__ W2t,
    const float* __restrict__ aw_s, const float* __restrict__ aw_d,
    float* __restrict__ h2, float* __restrict__ as2, float* __restrict__ ad2, int N)
{
  int wave = (int)(((unsigned)blockIdx.x * blockDim.x + threadIdx.x) >> 5);
  int lane = threadIdx.x & 31;
  int mtiles = (N + 15) >> 4;
  if (wave >= mtiles) return;
  int half = lane >> 4, l15 = lane & 15;
  int mbase = wave << 4;
  int arow = mbase + l15; if (arow > N - 1) arow = N - 1;
  const float* ap = hin + (size_t)arow * 64 + half * 16;

  v8f acc0 = {}, acc1 = {}, acc2 = {};
  #pragma unroll
  for (int k = 0; k < 64; k += 32) {
    const float* pa = ap + k;
    float4 f0 = ((const float4*)pa)[0];
    float4 f1 = ((const float4*)pa)[1];
    float4 f2 = ((const float4*)(pa + 8))[0];
    float4 f3 = ((const float4*)(pa + 8))[1];
    v16h a;
    a[0]=(_Float16)f0.x;  a[1]=(_Float16)f0.y;  a[2]=(_Float16)f0.z;  a[3]=(_Float16)f0.w;
    a[4]=(_Float16)f1.x;  a[5]=(_Float16)f1.y;  a[6]=(_Float16)f1.z;  a[7]=(_Float16)f1.w;
    a[8]=(_Float16)f2.x;  a[9]=(_Float16)f2.y;  a[10]=(_Float16)f2.z; a[11]=(_Float16)f2.w;
    a[12]=(_Float16)f3.x; a[13]=(_Float16)f3.y; a[14]=(_Float16)f3.z; a[15]=(_Float16)f3.w;
    const _Float16* pb = W2t + (size_t)l15 * 64 + k + half * 16;
    v16h b0 = *(const v16h*)(pb);
    v16h b1 = *(const v16h*)(pb + 16 * 64);
    v16h b2 = *(const v16h*)(pb + 32 * 64);
    acc0 = __builtin_amdgcn_wmma_f32_16x16x32_f16(false, a, false, b0, (short)0, acc0, false, false);
    acc1 = __builtin_amdgcn_wmma_f32_16x16x32_f16(false, a, false, b1, (short)0, acc1, false, false);
    acc2 = __builtin_amdgcn_wmma_f32_16x16x32_f16(false, a, false, b2, (short)0, acc2, false, false);
  }

  auto epi = [&](const v8f& A, int nb) {
    int col = nb + l15;
    bool cval = col < 40;
    float ws_ = cval ? aw_s[col] : 0.0f;
    float wd_ = cval ? aw_d[col] : 0.0f;
    #pragma unroll
    for (int r = 0; r < 8; ++r) {
      int row = mbase + r + half * 8;
      float v = A[r];
      bool rv = row < N;
      if (rv && cval) h2[(size_t)row * 40 + col] = v;
      float s = v * ws_;
      float d = v * wd_;
      s += __shfl_xor(s, 1); s += __shfl_xor(s, 2); s += __shfl_xor(s, 4); s += __shfl_xor(s, 8);
      d += __shfl_xor(d, 1); d += __shfl_xor(d, 2); d += __shfl_xor(d, 4); d += __shfl_xor(d, 8);
      if (rv && l15 == 0) { atomicAdd(&as2[row], s); atomicAdd(&ad2[row], d); }
    }
  };
  epi(acc0, 0); epi(acc1, 16); epi(acc2, 32);
}

// ---------------------------------------------------------------- layer-2 edge passes (H=1, C=40)
__global__ void k_edge_max2(const int* __restrict__ src, const int* __restrict__ dst,
                            const float* __restrict__ as2, const float* __restrict__ ad2,
                            float* __restrict__ mx2, int E, int Ne) {
  int e = blockIdx.x * blockDim.x + threadIdx.x;
  if (e >= Ne) return;
  int s, d; if (e < E) { s = src[e]; d = dst[e]; } else { s = e - E; d = s; }
  atomicMaxFloat(mx2 + d, lrelu(as2[s] + ad2[d]));
}

__global__ void k_edge_den2(const int* __restrict__ src, const int* __restrict__ dst,
                            const float* __restrict__ as2, const float* __restrict__ ad2,
                            const float* __restrict__ mx2, float* __restrict__ dn2,
                            int E, int Ne) {
  int e = blockIdx.x * blockDim.x + threadIdx.x;
  if (e >= Ne) return;
  int s, d; if (e < E) { s = src[e]; d = dst[e]; } else { s = e - E; d = s; }
  atomicAdd(dn2 + d, __expf(lrelu(as2[s] + ad2[d]) - mx2[d]));
}

// one thread per (edge, 5-col chunk): 8 chunks cover 40 cols
__global__ void k_edge_msg2(const int* __restrict__ src, const int* __restrict__ dst,
                            const float* __restrict__ as2, const float* __restrict__ ad2,
                            const float* __restrict__ mx2, const float* __restrict__ dn2,
                            const float* __restrict__ h2, float* __restrict__ out,
                            int E, int Ne) {
  int tid = blockIdx.x * blockDim.x + threadIdx.x;
  int e = tid >> 3, r = tid & 7;
  if (e >= Ne) return;
  int s, d; if (e < E) { s = src[e]; d = dst[e]; } else { s = e - E; d = s; }
  float v = lrelu(as2[s] + ad2[d]);
  float alpha = __expf(v - mx2[d]) / (dn2[d] + 1e-16f);
  const float* hp = h2 + (size_t)s * 40 + r * 5;
  float* op = out + (size_t)d * 40 + r * 5;
  #pragma unroll
  for (int c = 0; c < 5; ++c) atomicAdd(op + c, alpha * hp[c]);
}

__global__ void k_bias2(float* __restrict__ out, const float* __restrict__ b2, int N) {
  int i = blockIdx.x * blockDim.x + threadIdx.x;
  if (i >= N * 40) return;
  out[i] += b2[i % 40];
}

// ---------------------------------------------------------------- launch
extern "C" void kernel_launch(void* const* d_in, const int* in_sizes, int n_in,
                              void* d_out, int out_size, void* d_ws, size_t ws_size,
                              hipStream_t stream) {
  const float* x   = (const float*)d_in[0];
  const int*   ei  = (const int*)d_in[1];
  const float* W1  = (const float*)d_in[2];
  const float* aS1 = (const float*)d_in[3];
  const float* aD1 = (const float*)d_in[4];
  const float* b1  = (const float*)d_in[5];
  const float* W2  = (const float*)d_in[6];
  const float* aS2 = (const float*)d_in[7];
  const float* aD2 = (const float*)d_in[8];
  const float* b2  = (const float*)d_in[9];

  int N  = in_sizes[0] / 512;
  int E  = in_sizes[1] / 2;
  int Ne = E + N;
  const int* srcI = ei;
  const int* dstI = ei + E;

  float* ws  = (float*)d_ws;
  size_t off = 0;
  float* h1  = ws + off; off += (size_t)N * 64;
  float* as1 = ws + off; off += (size_t)N * 8;
  float* ad1 = ws + off; off += (size_t)N * 8;
  float* mx1 = ws + off; off += (size_t)N * 8;
  float* dn1 = ws + off; off += (size_t)N * 8;
  float* o1  = ws + off; off += (size_t)N * 64;
  float* h2  = ws + off; off += (size_t)N * 40;
  float* as2 = ws + off; off += (size_t)N;
  float* ad2 = ws + off; off += (size_t)N;
  float* mx2 = ws + off; off += (size_t)N;
  float* dn2 = ws + off; off += (size_t)N;
  _Float16* W1t = (_Float16*)(ws + off); off += (size_t)(64 * 512) / 2;
  _Float16* W2t = (_Float16*)(ws + off); off += (size_t)(48 * 64) / 2;
  float* out = (float*)d_out;

  const int B = 256;
  int mtiles = (N + 15) / 16;
  long long tmsg = (long long)Ne * 8;

  k_init<<<(N * 64 + B - 1) / B, B, 0, stream>>>(o1, mx1, dn1, mx2, dn2, as2, ad2, out, N);
  k_prep_w<<<(64 * 512 + B - 1) / B, B, 0, stream>>>(W1, W2, W1t, W2t);
  k_gemm1<<<(mtiles * 32 + B - 1) / B, B, 0, stream>>>(x, W1t, aS1, aD1, h1, as1, ad1, N);
  k_edge_max1<<<(Ne + B - 1) / B, B, 0, stream>>>(srcI, dstI, as1, ad1, mx1, E, Ne);
  k_edge_den1<<<(Ne + B - 1) / B, B, 0, stream>>>(srcI, dstI, as1, ad1, mx1, dn1, E, Ne);
  k_edge_msg1<<<(int)((tmsg + B - 1) / B), B, 0, stream>>>(srcI, dstI, as1, ad1, mx1, dn1, h1, o1, E, Ne);
  k_bias_elu<<<(N * 64 + B - 1) / B, B, 0, stream>>>(o1, b1, N);
  k_gemm2<<<(mtiles * 32 + B - 1) / B, B, 0, stream>>>(o1, W2t, aS2, aD2, h2, as2, ad2, N);
  k_edge_max2<<<(Ne + B - 1) / B, B, 0, stream>>>(srcI, dstI, as2, ad2, mx2, E, Ne);
  k_edge_den2<<<(Ne + B - 1) / B, B, 0, stream>>>(srcI, dstI, as2, ad2, mx2, dn2, E, Ne);
  k_edge_msg2<<<(int)((tmsg + B - 1) / B), B, 0, stream>>>(srcI, dstI, as2, ad2, mx2, dn2, h2, out, E, Ne);
  k_bias2<<<(N * 40 + B - 1) / B, B, 0, stream>>>(out, b2, N);
}
